// SMPLLayer_44203803410576
// MI455X (gfx1250) — matrix operand order repaired and checked by reading, not compiled, not run on hardware
//
#include <hip/hip_runtime.h>
#include <hip/hip_bf16.h>
#include <math.h>

typedef __attribute__((ext_vector_type(16))) _Float16 v16h;
typedef __attribute__((ext_vector_type(8)))  float    v8f;
typedef __attribute__((ext_vector_type(4)))  unsigned u32x4;
typedef __attribute__((ext_vector_type(8)))  int      i32x8;
typedef __attribute__((ext_vector_type(4)))  int      i32x4;

union V16H { v16h v; _Float16 s[16]; unsigned u[8]; };

#define VNUM   6890
#define C3     20670          // V*3 (true row length)
#define C3P    20672          // padded v_shaped row (16B-aligned rows)
#define BATCH  1024
#define KJ     24
#define NB     10
#define NP     207
#define NPPAD  224

// d_out offsets (floats): vertices, joints_t, joints, rot_mats
#define OUT_V   0
#define OUT_JT  21166080
#define OUT_J   21239808
#define OUT_R   21313536

// workspace offsets (floats)
#define WS_VSHAPED 0
#define WS_PF      (BATCH * C3P)            // pose features stored as f16 here
#define WS_A       (WS_PF + (BATCH * NPPAD) / 2)

__device__ __constant__ int PAR[24] =
  {-1,0,0,0,1,2,3,4,5,6,7,8,9,9,9,12,13,14,16,17,18,19,20,21};

__device__ __forceinline__ int a_kmap(int e, int hi) {
  return ((e & 8) << 1) + hi * 8 + (e & 7);   // 16-bit A 16x32 element->K
}

// ---- CDNA5 async global->LDS copy (ASYNCcnt) -------------------------------
__device__ __forceinline__ void async_b128(void* lds, const float* g) {
  unsigned off = (unsigned)(size_t)lds;
  asm volatile("global_load_async_to_lds_b128 %0, %1, off"
               :: "v"(off), "v"(g) : "memory");
}
#define WAIT_ASYNC(n) asm volatile("s_wait_asynccnt " #n ::: "memory")
#define WAIT_DS0()    asm volatile("s_wait_dscnt 0x0" ::: "memory")
#define WAIT_TENSOR(n) __builtin_amdgcn_s_wait_tensorcnt(n)

// ---- CDNA5 Tensor Data Mover: 2D tile global->LDS --------------------------
// ds_code: log2(element bytes). Zero-fills beyond tensor_w (x) / tensor_h (y).
__device__ __forceinline__ void tdm_load_2d(
    void* lds, const void* gsrc, int ds_code,
    unsigned tensor_w, unsigned tensor_h,
    unsigned tile_w, unsigned tile_h,
    unsigned long long row_stride_elems)
{
  unsigned long long ga = (unsigned long long)(size_t)gsrc;
  u32x4 g0;
  g0[0] = 1u;                                   // count=1, user descriptor
  g0[1] = (unsigned)(size_t)lds;                // lds_addr (bytes)
  g0[2] = (unsigned)ga;                         // global_addr[31:0]
  g0[3] = (unsigned)(ga >> 32) | 0x80000000u;   // global_addr[56:32] | type=2
  i32x8 g1;
  g1[0] = ds_code << 16;                                        // data_size
  g1[1] = (int)((tensor_w & 0xFFFFu) << 16);                    // dim0[15:0]
  g1[2] = (int)((tensor_w >> 16) | ((tensor_h & 0xFFFFu) << 16));
  g1[3] = (int)((tensor_h >> 16) | (tile_w << 16));             // tile_dim0
  g1[4] = (int)tile_h;                                          // tile_dim1 (tile_dim2=0)
  g1[5] = (int)(row_stride_elems & 0xFFFFFFFFull);              // dim0_stride lo
  g1[6] = (int)(row_stride_elems >> 32);                        // dim0_stride hi
  g1[7] = 0;
  i32x4 gz = {0, 0, 0, 0};
#if defined(__clang_major__) && (__clang_major__ >= 23)
  i32x8 gz8 = {0, 0, 0, 0, 0, 0, 0, 0};
  __builtin_amdgcn_tensor_load_to_lds(g0, g1, gz, gz, gz8, 0);
#else
  __builtin_amdgcn_tensor_load_to_lds(g0, g1, gz, gz, 0);
#endif
}

// ---------------------------------------------------------------------------
// Kernel 1: v_shaped = v_template + betas @ shapedirs^T   (M=c, N=b, K=10->32)
// betas tile async-staged to LDS (ASYNCcnt path); branchless fragment builds.
// ---------------------------------------------------------------------------
__global__ __launch_bounds__(32) void k_shape_blend(
    const float* __restrict__ betas, const float* __restrict__ shapedirs,
    const float* __restrict__ v_template, float* __restrict__ vshaped)
{
  __shared__ float sbet[192];                 // 16 x 10 betas tile (+pad)
  const int lane = threadIdx.x;
  const int c0 = blockIdx.x * 16;
  const int b0 = blockIdx.y * 16;
  const int m = lane & 15, hi = lane >> 4;
  const int c = c0 + m;
  const bool tail = (c0 + 16 > C3);

#pragma unroll
  for (int it = 0; it < 2; ++it) {            // 160 floats = 40 b128 chunks
    int i = it * 32 + lane;
    if (i < 40)
      async_b128(&sbet[i * 4], betas + (size_t)b0 * NB + i * 4);
  }

  float vals[10];
  if (!tail) {
    const float* r = shapedirs + (size_t)c * NB;
#pragma unroll
    for (int i = 0; i < 5; ++i) {
      float2 q = *(const float2*)(r + 2 * i);
      vals[2 * i] = q.x; vals[2 * i + 1] = q.y;
    }
  } else {
#pragma unroll
    for (int l = 0; l < 10; ++l)
      vals[l] = (c < C3) ? shapedirs[(size_t)c * NB + l] : 0.0f;
  }
  V16H a;
#pragma unroll
  for (int e = 0; e < 16; ++e) a.s[e] = (_Float16)0.0f;
  if (hi == 0) {
#pragma unroll
    for (int e = 0; e < 8; ++e) a.s[e] = (_Float16)vals[e];
  } else {
    a.s[0] = (_Float16)vals[8];
    a.s[1] = (_Float16)vals[9];
  }
  WAIT_ASYNC(0);
  // branchless: clamp index so LDS load is unconditional, then cndmask zero
  V16H b;
#pragma unroll
  for (int e = 0; e < 16; ++e) {
    const int  kk  = a_kmap(e, hi);
    const bool kok = (kk < NB);
    float x = sbet[m * NB + (kok ? kk : 0)];
    b.s[e] = kok ? (_Float16)x : (_Float16)0.0f;
  }

  v8f acc = {};
  acc = __builtin_amdgcn_wmma_f32_16x16x32_f16(false, a.v, false, b.v,
                                               (short)0, acc, false, false);
  const int bb = b0 + m;
  const int cbase = c0 + 8 * hi;
  float4 vt0, vt1;
  if (!tail) {
    vt0 = *(const float4*)(v_template + cbase);
    vt1 = *(const float4*)(v_template + cbase + 4);
  } else {
    float t[8];
#pragma unroll
    for (int r = 0; r < 8; ++r) t[r] = (cbase + r < C3) ? v_template[cbase + r] : 0.0f;
    vt0 = make_float4(t[0], t[1], t[2], t[3]);
    vt1 = make_float4(t[4], t[5], t[6], t[7]);
  }
  float4 o0 = make_float4(acc[0] + vt0.x, acc[1] + vt0.y, acc[2] + vt0.z, acc[3] + vt0.w);
  float4 o1 = make_float4(acc[4] + vt1.x, acc[5] + vt1.y, acc[6] + vt1.z, acc[7] + vt1.w);
  float* dst = vshaped + (size_t)bb * C3P + cbase;
  *(float4*)dst = o0;
  *(float4*)(dst + 4) = o1;
}

// ---------------------------------------------------------------------------
// Kernel 2: joints_t (all 3 comps) = Jreg @ v_shaped.  M=j, N=b, K=v.
// Double-buffered TDM staging of 16x96 v_shaped slabs (TENSORcnt ping-pong).
// ---------------------------------------------------------------------------
__global__ __launch_bounds__(32) void k_joints(
    const float* __restrict__ Jreg, const float* __restrict__ vsh,
    float* __restrict__ out)
{
  __shared__ float sb[2][16][96];
  const int lane = threadIdx.x;
  const int b0 = blockIdx.x * 16;
  const int j0 = blockIdx.y * 16;
  const int m = lane & 15, hi = lane >> 4;
  const int j = j0 + m;
  const bool jok = (j < KJ);

  v8f acc0 = {}, acc1 = {}, acc2 = {};

  auto stage = [&](int buf, int kb) {         // one TDM op per 16x96 slab
    tdm_load_2d(&sb[buf][0][0], vsh + (size_t)b0 * C3P + kb * 3,
                /*ds=*/2, /*tw=*/96, /*th=*/16, /*tile_w=*/96, /*tile_h=*/16,
                /*stride=*/C3P);
  };

  stage(0, 0);
  const int NSTEP = 215;                      // kb = 0..6848 fully in-range
  for (int step = 0; step < NSTEP; ++step) {
    const int kb = step * 32;
    const int cur = step & 1;
    if (step + 1 < NSTEP) {
      WAIT_DS0();                             // LDS reads done before overwrite
      stage(cur ^ 1, kb + 32);
      WAIT_TENSOR(1);                         // previous slab complete (in-order)
    } else {
      WAIT_TENSOR(0);
    }
    V16H a;
    if (jok) {
      const float* row = Jreg + (size_t)j * VNUM + kb + hi * 8;
#pragma unroll
      for (int i = 0; i < 4; ++i) {
        float2 lo = *(const float2*)(row + 2 * i);
        float2 hv = *(const float2*)(row + 16 + 2 * i);
        a.s[2 * i]     = (_Float16)lo.x; a.s[2 * i + 1] = (_Float16)lo.y;
        a.s[8 + 2 * i] = (_Float16)hv.x; a.s[9 + 2 * i] = (_Float16)hv.y;
      }
      if (kb + 64 < VNUM) __builtin_prefetch(row + 32, 0, 1);
    } else {
#pragma unroll
      for (int e = 0; e < 16; ++e) a.s[e] = (_Float16)0.0f;
    }
    V16H f0, f1, f2;
#pragma unroll
    for (int e = 0; e < 16; ++e) {
      const int k3 = a_kmap(e, hi) * 3;
      f0.s[e] = (_Float16)sb[cur][m][k3 + 0];
      f1.s[e] = (_Float16)sb[cur][m][k3 + 1];
      f2.s[e] = (_Float16)sb[cur][m][k3 + 2];
    }
    acc0 = __builtin_amdgcn_wmma_f32_16x16x32_f16(false, a.v, false, f0.v, (short)0, acc0, false, false);
    acc1 = __builtin_amdgcn_wmma_f32_16x16x32_f16(false, a.v, false, f1.v, (short)0, acc1, false, false);
    acc2 = __builtin_amdgcn_wmma_f32_16x16x32_f16(false, a.v, false, f2.v, (short)0, acc2, false, false);
  }
  {                                           // tail kb = 6880 (10 valid v)
    const int kb = 6880;
    V16H a;
#pragma unroll
    for (int e = 0; e < 16; ++e) {
      int k = kb + a_kmap(e, hi);
      a.s[e] = (jok && k < VNUM) ? (_Float16)Jreg[(size_t)j * VNUM + k] : (_Float16)0.0f;
    }
    V16H f0, f1, f2;
#pragma unroll
    for (int e = 0; e < 16; ++e) {
      const int kv = kb + a_kmap(e, hi);
      const bool kvok = (kv < VNUM);
      size_t base = (size_t)(b0 + m) * C3P + kv * 3;
      f0.s[e] = kvok ? (_Float16)vsh[base + 0] : (_Float16)0.0f;
      f1.s[e] = kvok ? (_Float16)vsh[base + 1] : (_Float16)0.0f;
      f2.s[e] = kvok ? (_Float16)vsh[base + 2] : (_Float16)0.0f;
    }
    acc0 = __builtin_amdgcn_wmma_f32_16x16x32_f16(false, a.v, false, f0.v, (short)0, acc0, false, false);
    acc1 = __builtin_amdgcn_wmma_f32_16x16x32_f16(false, a.v, false, f1.v, (short)0, acc1, false, false);
    acc2 = __builtin_amdgcn_wmma_f32_16x16x32_f16(false, a.v, false, f2.v, (short)0, acc2, false, false);
  }
  const int bb = b0 + m;
#pragma unroll
  for (int r = 0; r < 8; ++r) {
    int jj = j0 + r + 8 * hi;
    if (jj < KJ) {
      out[OUT_JT + bb * 72 + jj * 3 + 0] = acc0[r];
      out[OUT_JT + bb * 72 + jj * 3 + 1] = acc1[r];
      out[OUT_JT + bb * 72 + jj * 3 + 2] = acc2[r];
    }
  }
}

// ---------------------------------------------------------------------------
// Kernel 3: Rodrigues, pose features (f16-packed), chain, A mats.
// ---------------------------------------------------------------------------
__global__ __launch_bounds__(32) void k_pose_chain(
    const float* __restrict__ body_pose, const float* __restrict__ global_orient,
    float* __restrict__ out, float* __restrict__ ws)
{
  __shared__ float R[24][9];
  __shared__ float rel[24][3];
  __shared__ float jt[24][3];
  __shared__ float G[24][12];
  const int b = blockIdx.x;
  const int j = threadIdx.x;
  _Float16* pf16 = (_Float16*)(ws + WS_PF);

  if (j < KJ) {
    float a0, a1, a2;
    if (j == 0) {
      a0 = global_orient[b * 3 + 0];
      a1 = global_orient[b * 3 + 1];
      a2 = global_orient[b * 3 + 2];
    } else {
      const float* p = body_pose + b * 69 + (j - 1) * 3;
      a0 = p[0]; a1 = p[1]; a2 = p[2];
    }
    float x = a0 + 1e-8f, y = a1 + 1e-8f, z = a2 + 1e-8f;
    float ang = sqrtf(x * x + y * y + z * z);
    float inv = 1.0f / ang;
    float rx = a0 * inv, ry = a1 * inv, rz = a2 * inv;
    float s = sinf(ang), oc = 1.0f - cosf(ang);
    float Rm[9];
    Rm[0] = 1.0f - oc * (ry * ry + rz * rz);
    Rm[1] = -s * rz + oc * rx * ry;
    Rm[2] =  s * ry + oc * rx * rz;
    Rm[3] =  s * rz + oc * rx * ry;
    Rm[4] = 1.0f - oc * (rx * rx + rz * rz);
    Rm[5] = -s * rx + oc * ry * rz;
    Rm[6] = -s * ry + oc * rx * rz;
    Rm[7] =  s * rx + oc * ry * rz;
    Rm[8] = 1.0f - oc * (rx * rx + ry * ry);
#pragma unroll
    for (int i = 0; i < 9; ++i) {
      R[j][i] = Rm[i];
      out[OUT_R + b * 216 + j * 9 + i] = Rm[i];
    }
    if (j > 0) {
      int base = b * NPPAD + (j - 1) * 9;
#pragma unroll
      for (int i = 0; i < 9; ++i)
        pf16[base + i] = (_Float16)(Rm[i] - ((i == 0 || i == 4 || i == 8) ? 1.0f : 0.0f));
    }
#pragma unroll
    for (int c = 0; c < 3; ++c)
      jt[j][c] = out[OUT_JT + b * 72 + j * 3 + c];
  }
  if (j == 0)
    for (int p = NP; p < NPPAD; ++p) pf16[b * NPPAD + p] = (_Float16)0.0f;
  __syncthreads();
  if (j < KJ) {
#pragma unroll
    for (int c = 0; c < 3; ++c)
      rel[j][c] = (j == 0) ? jt[0][c] : (jt[j][c] - jt[PAR[j]][c]);
  }
  __syncthreads();
  if (j == 0) {
    for (int i = 0; i < 3; ++i) {
      for (int c = 0; c < 3; ++c) G[0][i * 4 + c] = R[0][i * 3 + c];
      G[0][i * 4 + 3] = rel[0][i];
    }
    for (int k = 1; k < KJ; ++k) {
      int p = PAR[k];
      for (int i = 0; i < 3; ++i) {
        for (int c = 0; c < 3; ++c)
          G[k][i * 4 + c] = G[p][i * 4 + 0] * R[k][0 + c] +
                            G[p][i * 4 + 1] * R[k][3 + c] +
                            G[p][i * 4 + 2] * R[k][6 + c];
        G[k][i * 4 + 3] = G[p][i * 4 + 0] * rel[k][0] +
                          G[p][i * 4 + 1] * rel[k][1] +
                          G[p][i * 4 + 2] * rel[k][2] + G[p][i * 4 + 3];
      }
    }
    for (int k = 0; k < KJ; ++k)
      for (int c = 0; c < 3; ++c)
        out[OUT_J + b * 72 + k * 3 + c] = G[k][c * 4 + 3];
  }
  __syncthreads();
  if (j < KJ) {
    float* A = ws + WS_A + (size_t)(b * KJ + j) * 12;
#pragma unroll
    for (int i = 0; i < 3; ++i) {
      float d = G[j][i * 4 + 0] * jt[j][0] + G[j][i * 4 + 1] * jt[j][1] +
                G[j][i * 4 + 2] * jt[j][2];
      A[i * 4 + 0] = G[j][i * 4 + 0];
      A[i * 4 + 1] = G[j][i * 4 + 1];
      A[i * 4 + 2] = G[j][i * 4 + 2];
      A[i * 4 + 3] = G[j][i * 4 + 3] - d;
    }
  }
}

// ---------------------------------------------------------------------------
// Kernel 4: fused pose-blend WMMA + LBS skinning.  16 batches x 16 vertices.
// Four TDM tile loads; tile_h=NPPAD on the posedirs slab makes the TDM
// zero-fill rows 207..223 in hardware -> unconditional B-fragment loads.
// ---------------------------------------------------------------------------
__global__ __launch_bounds__(32) void k_skin(
    const float* __restrict__ posedirs, const float* __restrict__ lbsw,
    const float* __restrict__ wsr, float* __restrict__ out)
{
  __shared__ float    Bst[NPPAD][48];   // posedirs slab + HW-zeroed K padding
  __shared__ _Float16 pfs[16][NPPAD];   // pose-feature slab (7 KB)
  __shared__ float    Am[16][288];
  __shared__ float    wl[16][24];
  __shared__ float    vp[16][48];
  const int lane = threadIdx.x;
  const int v0 = blockIdx.x * 16;
  const int b0 = blockIdx.y * 16;
  const int m = lane & 15, hi = lane >> 4;
  const int c0 = v0 * 3;

  const _Float16* pf16 = (const _Float16*)(wsr + WS_PF);
  const float* vsh = wsr + WS_VSHAPED;
  const float* Aw  = wsr + WS_A;

  tdm_load_2d(&Bst[0][0], posedirs + c0, 2,
              (unsigned)(C3 - c0), NP, 48, NPPAD, C3);       // x & y clamped
  tdm_load_2d(&pfs[0][0], pf16 + (size_t)b0 * NPPAD, 1,
              NPPAD, 16, NPPAD, 16, NPPAD);
  tdm_load_2d(&Am[0][0], Aw + (size_t)b0 * 288, 2,
              288, 16, 288, 16, 288);
  tdm_load_2d(&wl[0][0], lbsw + (size_t)v0 * 24, 2,
              24, (unsigned)(VNUM - v0), 24, 16, 24);        // y-clamped tail

  WAIT_TENSOR(2);                        // Bst + pfs landed (in-order TDM)

  for (int t = 0; t < 3; ++t) {
    v8f acc = {};
    for (int kb = 0; kb < NPPAD; kb += 32) {
      V16H a, b;
#pragma unroll
      for (int i = 0; i < 8; ++i) {      // f16-pair dwords, exact A layout
        int k0 = kb + hi * 8 + ((i < 4) ? (2 * i) : (2 * i + 8));
        a.u[i] = *(const unsigned*)&pfs[m][k0];
      }
      // rows >= NP are HW-zeroed: no guard
      const float* br = &Bst[kb][t * 16 + m];
#pragma unroll
      for (int e = 0; e < 16; ++e)
        b.s[e] = (_Float16)br[a_kmap(e, hi) * 48];
      acc = __builtin_amdgcn_wmma_f32_16x16x32_f16(false, a.v, false, b.v,
                                                   (short)0, acc, false, false);
    }
#pragma unroll
    for (int r = 0; r < 8; ++r)
      vp[r + 8 * hi][t * 16 + m] = acc[r];
  }
  WAIT_TENSOR(0);                        // Am + wl landed
  __syncthreads();

  for (int it = 0; it < 8; ++it) {
    int p = it * 32 + lane;
    int bi = p >> 4, vi = p & 15;
    int v = v0 + vi;
    if (v < VNUM) {
      size_t sbase = (size_t)(b0 + bi) * C3P + v * 3;
      size_t obase = (size_t)(b0 + bi) * C3  + v * 3;
      float x = vp[bi][vi * 3 + 0] + vsh[sbase + 0];
      float y = vp[bi][vi * 3 + 1] + vsh[sbase + 1];
      float z = vp[bi][vi * 3 + 2] + vsh[sbase + 2];
      float o0 = 0.f, o1 = 0.f, o2 = 0.f;
#pragma unroll 4
      for (int k = 0; k < 24; ++k) {
        float wk = wl[vi][k];
        float4 A0 = *(const float4*)&Am[bi][k * 12 + 0];
        float4 A1 = *(const float4*)&Am[bi][k * 12 + 4];
        float4 A2 = *(const float4*)&Am[bi][k * 12 + 8];
        o0 += wk * (A0.x * x + A0.y * y + A0.z * z + A0.w);
        o1 += wk * (A1.x * x + A1.y * y + A1.z * z + A1.w);
        o2 += wk * (A2.x * x + A2.y * y + A2.z * z + A2.w);
      }
      out[OUT_V + obase + 0] = o0;
      out[OUT_V + obase + 1] = o1;
      out[OUT_V + obase + 2] = o2;
    }
  }
}

extern "C" void kernel_launch(void* const* d_in, const int* in_sizes, int n_in,
                              void* d_out, int out_size, void* d_ws, size_t ws_size,
                              hipStream_t stream) {
  (void)in_sizes; (void)n_in; (void)out_size; (void)ws_size;
  const float* body_pose     = (const float*)d_in[0];
  const float* betas         = (const float*)d_in[1];
  const float* global_orient = (const float*)d_in[2];
  const float* v_template    = (const float*)d_in[3];
  const float* shapedirs     = (const float*)d_in[4];
  const float* posedirs      = (const float*)d_in[5];
  const float* J_regressor   = (const float*)d_in[6];
  const float* lbs_weights   = (const float*)d_in[7];
  float* out = (float*)d_out;
  float* ws  = (float*)d_ws;

  dim3 g1((C3 + 15) / 16, BATCH / 16);
  k_shape_blend<<<g1, 32, 0, stream>>>(betas, shapedirs, v_template, ws + WS_VSHAPED);

  dim3 g2(BATCH / 16, 2);
  k_joints<<<g2, 32, 0, stream>>>(J_regressor, ws + WS_VSHAPED, out);

  k_pose_chain<<<dim3(BATCH), 32, 0, stream>>>(body_pose, global_orient, out, ws);

  dim3 g4((VNUM + 15) / 16, BATCH / 16);
  k_skin<<<g4, 32, 0, stream>>>(posedirs, lbs_weights, ws, out);
}
